// NerfWGarfieldLoss_72928544686695
// MI455X (gfx1250) — compile-verified, matrix operand order checked by reading, not run on hardware
//
#include <hip/hip_runtime.h>

typedef __attribute__((ext_vector_type(16))) _Float16 v16h;
typedef __attribute__((ext_vector_type(8)))  _Float16 v8h;
typedef __attribute__((ext_vector_type(8)))  float    v8f;
typedef __attribute__((ext_vector_type(4)))  float    v4f;

#define N_PTS 4096
#define S_DIM 64
#define D_DIM 64
#define H_DIM 256
#define F_DIM 256
#define N_CHUNKS 16   // 4096 / 256

// ---------------------------------------------------------------------------
// WMMA fragment loader (wave32, 16x16x32 f16).
// A-matrix 16x32: lane&15 selects M-row; lane half selects K offset {0,8};
// element i<8 -> k = off+i, i>=8 -> k = off+16+(i-8). B mirrors with N on lanes.
// rowptr must point at the start of the (lane-selected) row; rows 16B aligned.
// ---------------------------------------------------------------------------
__device__ __forceinline__ v16h load_frag(const _Float16* rowptr, int lane, int kBase) {
    const _Float16* p = rowptr + kBase + ((lane & 16) ? 8 : 0);
    v8h lo = *(const v8h*)(p);
    v8h hi = *(const v8h*)(p + 16);
    v16h f;
#pragma unroll
    for (int i = 0; i < 8; ++i) { f[i] = lo[i]; f[i + 8] = hi[i]; }
    return f;
}

__device__ __forceinline__ float block_reduce(float v, float* red) {
    int tid = threadIdx.x;
    red[tid] = v;
    __syncthreads();
    for (int s = blockDim.x >> 1; s > 0; s >>= 1) {
        if (tid < s) red[tid] += red[tid + s];
        __syncthreads();
    }
    float r = red[0];
    __syncthreads();
    return r;
}

// ---------------------------------------------------------------------------
// K0: zero the accumulator block in workspace
// acc[0]=msum acc[1]=c acc[2]=f acc[3]=logbeta acc[4]=sigma
// acc[5]=loss1 acc[6]=loss2 acc[7]=loss4 acc[8]=count
// ---------------------------------------------------------------------------
__global__ void k0_init(float* acc) {
    if (threadIdx.x < 16) acc[threadIdx.x] = 0.0f;
}

// ---------------------------------------------------------------------------
// K1: scalar losses (c_l, f_l, b_l, s_l partial sums)
// ---------------------------------------------------------------------------
__global__ void k1_scalar(const float* __restrict__ rgbs, const float* __restrict__ mask,
                          const float* __restrict__ rc,   const float* __restrict__ rf,
                          const float* __restrict__ beta, const float* __restrict__ sig,
                          float* acc) {
    float msum = 0.f, cs = 0.f, fs = 0.f, lb = 0.f, ss = 0.f;
    int tid = blockIdx.x * blockDim.x + threadIdx.x;
    int stride = gridDim.x * blockDim.x;
    for (int n = tid; n < N_PTS; n += stride) {
        float m = mask[n];
        msum += m;
        float b = beta[n];
        float inv2b2 = 1.0f / (2.0f * b * b);
        float c = 0.f, f = 0.f;
#pragma unroll
        for (int k = 0; k < 3; ++k) {
            float d1 = rc[n * 3 + k] - rgbs[n * 3 + k]; c += d1 * d1;
            float d2 = rf[n * 3 + k] - rgbs[n * 3 + k]; f += d2 * d2;
        }
        cs += c * m;
        fs += f * inv2b2 * m;
        lb += logf(b) * m;
    }
    for (int i = tid; i < N_PTS * S_DIM; i += stride) ss += sig[i];

    __shared__ float red[256];
    float r;
    r = block_reduce(msum, red); if (threadIdx.x == 0) atomicAdd(&acc[0], r);
    r = block_reduce(cs,   red); if (threadIdx.x == 0) atomicAdd(&acc[1], r);
    r = block_reduce(fs,   red); if (threadIdx.x == 0) atomicAdd(&acc[2], r);
    r = block_reduce(lb,   red); if (threadIdx.x == 0) atomicAdd(&acc[3], r);
    r = block_reduce(ss,   red); if (threadIdx.x == 0) atomicAdd(&acc[4], r);
}

// ---------------------------------------------------------------------------
// K2: feat[n][d] = sum_s enc[n][s][d] * w[n][s]
// 64 MB streaming read -> use 128-bit loads. 256 threads handle 16 rows:
// thread = (row r in 0..15, float4-column c in 0..15).
// ---------------------------------------------------------------------------
__global__ void k2_feat(const float* __restrict__ enc, const float* __restrict__ w,
                        _Float16* __restrict__ feat16) {
    const int rowBase = blockIdx.x * 16;
    const int tid = threadIdx.x;          // 256
    const int r = tid >> 4;               // local row 0..15
    const int c = tid & 15;               // float4 group 0..15 (d = 4c..4c+3)

    __shared__ float ws[16 * S_DIM];
    for (int i = tid; i < 16 * S_DIM; i += 256) ws[i] = w[rowBase * S_DIM + i];
    __syncthreads();

    const int n = rowBase + r;
    const v4f* e4 = (const v4f*)(enc + (size_t)n * S_DIM * D_DIM);
    float a0 = 0.f, a1 = 0.f, a2 = 0.f, a3 = 0.f;
#pragma unroll 8
    for (int s = 0; s < S_DIM; ++s) {
        v4f ev = e4[s * 16 + c];          // global_load_b128
        float wv = ws[r * S_DIM + s];
        a0 += ev[0] * wv; a1 += ev[1] * wv; a2 += ev[2] * wv; a3 += ev[3] * wv;
    }
    _Float16* o = feat16 + (size_t)n * D_DIM + c * 4;
    o[0] = (_Float16)a0; o[1] = (_Float16)a1; o[2] = (_Float16)a2; o[3] = (_Float16)a3;
}

// ---------------------------------------------------------------------------
// K2b: transpose-convert weights to f16 (B-fragments want contiguous K rows)
//   w1t[n*64 + k]  = W1[k][n]   (k<64; W1 row 64 used separately as bias)
//   w2t[n*256 + k] = W2[k][n]
// ---------------------------------------------------------------------------
__global__ void k2b_convert(const float* __restrict__ W1, const float* __restrict__ W2,
                            _Float16* __restrict__ w1t, _Float16* __restrict__ w2t) {
    const int total1 = H_DIM * D_DIM;     // 256*64
    const int total2 = F_DIM * H_DIM;     // 256*256
    int tid = blockIdx.x * blockDim.x + threadIdx.x;
    int stride = gridDim.x * blockDim.x;
    for (int i = tid; i < total1 + total2; i += stride) {
        if (i < total1) {
            int n = i >> 6, k = i & 63;
            w1t[i] = (_Float16)W1[k * H_DIM + n];
        } else {
            int j = i - total1;
            int n = j >> 8, k = j & 255;
            w2t[j] = (_Float16)W2[k * F_DIM + n];
        }
    }
}

// ---------------------------------------------------------------------------
// K3: fused MLP per 16-row tile:  h = relu(feat@W1top + s*W1[64]) ; g = h@W2
//     then row-normalize; outputs g16 (f16) + sq (f32) per variant.
// grid: (256 row-tiles, 2 variants); block: 128 threads = 4 waves,
// wave w owns output n-tiles {4w..4w+3}.
// ---------------------------------------------------------------------------
__global__ void k3_mlp(const _Float16* __restrict__ feat16,
                       const _Float16* __restrict__ w1t,
                       const float*    __restrict__ W1,      // for row 64 bias
                       const _Float16* __restrict__ w2t,
                       const float*    __restrict__ scales,
                       _Float16* __restrict__ g16, float* __restrict__ sq) {
    const int rowBase = blockIdx.x * 16;
    const int variant = blockIdx.y;
    const int tid  = threadIdx.x;
    const int lane = tid & 31;
    const int wave = tid >> 5;

    __shared__ _Float16 hsh[16 * H_DIM];   // 8 KB
    __shared__ float    gsh[16 * F_DIM];   // 16 KB
    __shared__ float    partial[128];
    __shared__ float    invs[16], sqv[16];

    const v8f zeroV = {};
    v8f acc1[4] = {zeroV, zeroV, zeroV, zeroV};

    // ---- GEMM1: (16 x 64) @ (64 x 256) ----
    const _Float16* arow1 = feat16 + (size_t)(rowBase + (lane & 15)) * D_DIM;
#pragma unroll
    for (int kb = 0; kb < D_DIM; kb += 32) {
        v16h a = load_frag(arow1, lane, kb);
#pragma unroll
        for (int t = 0; t < 4; ++t) {
            int ntile = wave * 4 + t;
            v16h b = load_frag(w1t + (size_t)(ntile * 16 + (lane & 15)) * D_DIM, lane, kb);
            acc1[t] = __builtin_amdgcn_wmma_f32_16x16x32_f16(false, a, false, b,
                                                             (short)0, acc1[t], false, false);
        }
    }

    // ---- bias (scale column of x) + ReLU -> LDS ----
    const int mOff = (lane & 16) ? 8 : 0;
    float sv[8];
#pragma unroll
    for (int r = 0; r < 8; ++r) {
        float s = scales[rowBase + r + mOff];
        if (variant) { float diff = fmaxf(0.0f, 2.0f - s); s = s + diff * 0.5f; }
        sv[r] = s;
    }
#pragma unroll
    for (int t = 0; t < 4; ++t) {
        int n = (wave * 4 + t) * 16 + (lane & 15);
        float w1last = W1[64 * H_DIM + n];
#pragma unroll
        for (int r = 0; r < 8; ++r) {
            float h = acc1[t][r] + sv[r] * w1last;
            hsh[(r + mOff) * H_DIM + n] = (_Float16)fmaxf(h, 0.0f);
        }
    }
    __syncthreads();

    // ---- GEMM2: (16 x 256) @ (256 x 256) ----
    v8f acc2[4] = {zeroV, zeroV, zeroV, zeroV};
    const _Float16* arow2 = hsh + (size_t)(lane & 15) * H_DIM;
#pragma unroll
    for (int kb = 0; kb < H_DIM; kb += 32) {
        v16h a = load_frag(arow2, lane, kb);
#pragma unroll
        for (int t = 0; t < 4; ++t) {
            int ntile = wave * 4 + t;
            v16h b = load_frag(w2t + (size_t)(ntile * 16 + (lane & 15)) * H_DIM, lane, kb);
            acc2[t] = __builtin_amdgcn_wmma_f32_16x16x32_f16(false, a, false, b,
                                                             (short)0, acc2[t], false, false);
        }
    }

    // ---- stash g_pre to LDS for row-normalization ----
#pragma unroll
    for (int t = 0; t < 4; ++t) {
        int n = (wave * 4 + t) * 16 + (lane & 15);
#pragma unroll
        for (int r = 0; r < 8; ++r) gsh[(r + mOff) * F_DIM + n] = acc2[t][r];
    }
    __syncthreads();

    // ---- parallel row-norm: 8 threads per row, then combine ----
    {
        int m = tid >> 3;                 // row 0..15
        int gcol = tid & 7;
        float s = 0.f;
        for (int n = gcol; n < F_DIM; n += 8) { float v = gsh[m * F_DIM + n]; s += v * v; }
        partial[tid] = s;
    }
    __syncthreads();
    if (tid < 16) {
        float sum = 0.f;
#pragma unroll
        for (int j = 0; j < 8; ++j) sum += partial[tid * 8 + j];
        float inv = 1.0f / (sqrtf(sum) + 1e-6f);
        invs[tid] = inv;
        sqv[tid]  = sum * inv * inv;      // ||g_norm||^2 as the reference computes it
    }
    __syncthreads();

    size_t gbase = (size_t)variant * N_PTS * F_DIM + (size_t)rowBase * F_DIM;
    for (int e = tid; e < 16 * F_DIM; e += blockDim.x) {
        int m = e >> 8;
        g16[gbase + e] = (_Float16)(gsh[e] * invs[m]);
    }
    if (tid < 16) sq[variant * N_PTS + rowBase + tid] = sqv[tid];
}

// ---------------------------------------------------------------------------
// K4: per-chunk pairwise distances via WMMA G@G^T, masked loss accumulation.
// The 8 KB A-tile (16 contiguous rows of g16) is staged into LDS with
// GLOBAL_LOAD_ASYNC_TO_LDS_B128 (ASYNCcnt path), then A-fragments come from
// LDS while B-fragments stream from global (chunk is L2-resident).
// grid: (16 chunks, 16 i-tiles, 2 variants); block 128 = 4 waves,
// wave w owns j-tiles {4w..4w+3}.
// variant 0 -> loss1, loss4, block_mask count; variant 1 -> loss2.
// ---------------------------------------------------------------------------
__global__ void k4_pairwise(const _Float16* __restrict__ g16, const float* __restrict__ sq,
                            const float* __restrict__ beta, const int* __restrict__ sam,
                            float* acc) {
    const int chunkId = blockIdx.x;
    const int itile   = blockIdx.y;
    const int variant = blockIdx.z;
    const int tid  = threadIdx.x;
    const int lane = tid & 31;
    const int wave = tid >> 5;

    const _Float16* g  = g16 + (size_t)variant * N_PTS * F_DIM;
    const float*    sv = sq + variant * N_PTS;
    const int chunkBase = chunkId * 256;
    const int rowI = chunkBase + itile * 16;

    __shared__ _Float16 ash[16 * F_DIM];   // 8 KB A tile
    __shared__ float    red[128];

    // ---- async-stage A tile: 16 rows x 256 f16 = contiguous 8192 B ----
    {
        const char* gsrc = (const char*)(g + (size_t)rowI * F_DIM);
        unsigned ldsBase = (unsigned)(uintptr_t)(&ash[0]);
#pragma unroll
        for (int it = 0; it < 4; ++it) {
            int byteOff = (it * 128 + tid) * 16;
            unsigned long long ga = (unsigned long long)(uintptr_t)(gsrc + byteOff);
            unsigned la = ldsBase + byteOff;
            asm volatile("global_load_async_to_lds_b128 %0, %1, off"
                         :: "v"(la), "v"(ga) : "memory");
        }
        asm volatile("s_wait_asynccnt 0x0" ::: "memory");
    }
    __syncthreads();

    const v8f zeroV = {};
    v8f accd[4] = {zeroV, zeroV, zeroV, zeroV};

    const _Float16* arow = ash + (size_t)(lane & 15) * F_DIM;
#pragma unroll
    for (int kb = 0; kb < F_DIM; kb += 32) {
        v16h a = load_frag(arow, lane, kb);
#pragma unroll
        for (int t = 0; t < 4; ++t) {
            int jtile = wave * 4 + t;
            v16h b = load_frag(g + (size_t)(chunkBase + jtile * 16 + (lane & 15)) * F_DIM,
                               lane, kb);
            accd[t] = __builtin_amdgcn_wmma_f32_16x16x32_f16(false, a, false, b,
                                                             (short)0, accd[t], false, false);
        }
    }

    const int mOff = (lane & 16) ? 8 : 0;
    int   liA[8], ilA[8];
    float biA[8], sqiA[8];
#pragma unroll
    for (int r = 0; r < 8; ++r) {
        int il = itile * 16 + r + mOff;
        int ig = chunkBase + il;
        ilA[r] = il; liA[r] = sam[ig]; biA[r] = beta[ig]; sqiA[r] = sv[ig];
    }

    float lpos = 0.f, lneg = 0.f, cnt = 0.f;
#pragma unroll
    for (int t = 0; t < 4; ++t) {
        int jl = (wave * 4 + t) * 16 + (lane & 15);
        int jg = chunkBase + jl;
        int   lj  = sam[jg];
        float bj  = beta[jg];
        float sqj = sv[jg];
        bool  sj  = bj < 0.15f;
#pragma unroll
        for (int r = 0; r < 8; ++r) {
            float d2 = sqiA[r] + sqj - 2.0f * accd[t][r];
            float d  = sqrtf(fmaxf(d2, 1e-12f));
            bool valid = (liA[r] != -1) && (lj != -1);
            bool stat  = (biA[r] < 0.15f) || sj;
            int  il = ilA[r];
            if (variant == 0) {
                if (valid && jl >= il) cnt += 1.0f;
                if (valid && stat && (liA[r] != lj) && (jl >= il))
                    lneg += fmaxf(0.0f, 1.0f - d);
            }
            if (valid && stat && (liA[r] == lj) && (jl > il)) lpos += d;
        }
    }

    float p = block_reduce(lpos, red);
    if (variant == 0) {
        float ng = block_reduce(lneg, red);
        float c  = block_reduce(cnt,  red);
        if (tid == 0) {
            atomicAdd(&acc[5], p);
            atomicAdd(&acc[7], ng);
            atomicAdd(&acc[8], c);
        }
    } else {
        if (tid == 0) atomicAdd(&acc[6], p);
    }
}

// ---------------------------------------------------------------------------
// K5: finalize the 5 outputs (COEF == ones)
// ---------------------------------------------------------------------------
__global__ void k5_finalize(const float* __restrict__ acc, float* __restrict__ out) {
    if (threadIdx.x == 0 && blockIdx.x == 0) {
        float msum = acc[0] + 1e-20f;
        out[0] = 0.5f * acc[1] / msum;
        out[1] = acc[2] / msum;
        out[2] = 3.0f + acc[3] / msum;
        out[3] = 0.01f * acc[4] / (float)(N_PTS * S_DIM);
        out[4] = (acc[5] + acc[6] + acc[7]) / acc[8];
    }
}

// ---------------------------------------------------------------------------
extern "C" void kernel_launch(void* const* d_in, const int* in_sizes, int n_in,
                              void* d_out, int out_size, void* d_ws, size_t ws_size,
                              hipStream_t stream) {
    const float* rgbs   = (const float*)d_in[0];
    const float* mask   = (const float*)d_in[1];
    const float* rc     = (const float*)d_in[2];
    const float* rf     = (const float*)d_in[3];
    const float* beta   = (const float*)d_in[4];
    const float* sig    = (const float*)d_in[5];
    const float* enc    = (const float*)d_in[6];
    const float* wfs    = (const float*)d_in[7];
    const float* scales = (const float*)d_in[8];
    const float* W1     = (const float*)d_in[9];
    const float* W2     = (const float*)d_in[10];
    const int*   sam    = (const int*)d_in[11];
    float* out = (float*)d_out;

    // workspace layout (256B-aligned blocks)
    char* ws = (char*)d_ws;
    float*    acc    = (float*)(ws + 0);                        //   64 B
    _Float16* feat16 = (_Float16*)(ws + 256);                   //  512 KB
    _Float16* w1t    = (_Float16*)(ws + 256 + 524288);          //   32 KB
    _Float16* w2t    = (_Float16*)(ws + 256 + 524288 + 32768);  //  128 KB
    float*    sq     = (float*)(ws + 256 + 524288 + 32768 + 131072);            // 32 KB
    _Float16* g16    = (_Float16*)(ws + 256 + 524288 + 32768 + 131072 + 32768); // 4 MB

    k0_init<<<1, 32, 0, stream>>>(acc);
    k1_scalar<<<64, 256, 0, stream>>>(rgbs, mask, rc, rf, beta, sig, acc);
    k2_feat<<<N_PTS / 16, 256, 0, stream>>>(enc, wfs, feat16);
    k2b_convert<<<320, 256, 0, stream>>>(W1, W2, w1t, w2t);
    k3_mlp<<<dim3(N_PTS / 16, 2), 128, 0, stream>>>(feat16, w1t, W1, w2t, scales, g16, sq);
    k4_pairwise<<<dim3(N_CHUNKS, 16, 2), 128, 0, stream>>>(g16, sq, beta, sam, acc);
    k5_finalize<<<1, 32, 0, stream>>>(acc, out);
}
